// TimeLSTMCell_14319420965206
// MI455X (gfx1250) — matrix operand order repaired and checked by reading, not compiled, not run on hardware
//
#include <hip/hip_runtime.h>
#include <hip/hip_bf16.h>

// Time-LSTM on MI455X (gfx1250): bf16 WMMA 16x16x32 with fp32 accumulation.
// Single fused kernel per scan step: each wave owns one 16x16 (batch,col)
// tile and carries FIVE WMMA accumulators (c_s1 + f,i,o,c_tmp gates) whose
// C-fragments share the same layout, so the whole LSTM cell update is
// register-local. Weights (3.5MB bf16) stay L2-resident across all 512 steps.

typedef __bf16 bf16_t;
typedef __attribute__((ext_vector_type(16))) __bf16 v16bf;
typedef __attribute__((ext_vector_type(8)))  __bf16 v8bf;
typedef __attribute__((ext_vector_type(8)))  float  v8f;

#define BB 64
#define SS 512
#define EE 256
#define HH 512
#define G4 2048   // 4*H

__device__ __forceinline__ v8f wmma_bf16(v16bf a, v16bf b, v8f c) {
  return __builtin_amdgcn_wmma_f32_16x16x32_bf16(
      /*neg_a=*/false, a, /*neg_b=*/false, b,
      /*c_mod=*/(short)0, c, /*reuse_a=*/false, /*reuse_b=*/false);
}

// A-fragment (16x32 bf16, M x K), row-major source with row stride ld.
// Lane l: row = l&15 ; elem e -> K = kk + 16*(l>=16) + e.
__device__ __forceinline__ v16bf load_a_frag(const bf16_t* __restrict__ mat,
                                             long ld, int m0, int kk, int lane) {
  const int row = m0 + (lane & 15);
  const int kb  = kk + ((lane >> 4) << 4);
  const bf16_t* p = mat + (long)row * ld + kb;
  v8bf lo = *(const v8bf*)(p);
  v8bf hi = *(const v8bf*)(p + 8);
  v16bf r;
#pragma unroll
  for (int e = 0; e < 8; ++e) { r[e] = lo[e]; r[e + 8] = hi[e]; }
  return r;
}

// B-fragment (32x16 bf16, K x N) where B[k][n] = W[n][k], W row-major (N rows,
// row stride ld).  Lane l: col = l&15 ; elem e -> K = kk + 16*(l>=16) + e.
// => each lane loads 16 contiguous bf16 (32 bytes) from one W row.
__device__ __forceinline__ v16bf load_b_frag(const bf16_t* __restrict__ w,
                                             long ld, int n0, int kk, int lane) {
  const int n  = n0 + (lane & 15);
  const int kb = kk + ((lane >> 4) << 4);
  return *(const v16bf*)(w + (long)n * ld + kb);
}

__device__ __forceinline__ float sigmoidf_(float x) {
  return 1.0f / (1.0f + __expf(-x));
}

// ---------------------------------------------------------------- setup ----
__global__ void tl_setup_kernel(const float* __restrict__ inputs,
                                const float* __restrict__ ts,
                                const float* __restrict__ W_all_w,
                                const float* __restrict__ W_all_b,
                                const float* __restrict__ U_w,
                                const float* __restrict__ U_b,
                                const float* __restrict__ W_d_w,
                                bf16_t* __restrict__ x_bf,
                                bf16_t* __restrict__ Wall_bf,
                                bf16_t* __restrict__ U_bf,
                                bf16_t* __restrict__ Wd_bf,
                                float* __restrict__ bias4,
                                float* __restrict__ decay,
                                bf16_t* __restrict__ h_bf0,
                                bf16_t* __restrict__ h_bf1,
                                bf16_t* __restrict__ c_bf0,
                                bf16_t* __restrict__ c_bf1,
                                float* __restrict__ c_f32) {
  const long tid = (long)blockIdx.x * blockDim.x + threadIdx.x;
  const long stride = (long)gridDim.x * blockDim.x;
  for (long i = tid; i < (long)BB * SS * EE; i += stride)
    x_bf[i] = (bf16_t)inputs[i];
  for (long i = tid; i < (long)G4 * HH; i += stride)
    Wall_bf[i] = (bf16_t)W_all_w[i];
  for (long i = tid; i < (long)G4 * EE; i += stride)
    U_bf[i] = (bf16_t)U_w[i];
  for (long i = tid; i < (long)HH * HH; i += stride)
    Wd_bf[i] = (bf16_t)W_d_w[i];
  for (long i = tid; i < G4; i += stride)
    bias4[i] = W_all_b[i] + U_b[i];
  for (long i = tid; i < (long)BB * SS; i += stride)
    decay[i] = 1.0f / __logf(2.718281828459045f + ts[i]);
  for (long i = tid; i < (long)BB * HH; i += stride) {
    h_bf0[i] = (bf16_t)0.0f;
    h_bf1[i] = (bf16_t)0.0f;
    c_bf0[i] = (bf16_t)0.0f;
    c_bf1[i] = (bf16_t)0.0f;
    c_f32[i] = 0.0f;
  }
}

// ----------------------------------------------------- fused scan step -----
// Grid: 16 blocks x 256 threads (8 waves). Block covers 64 rows x 32 cols;
// wave tile: rows m0 = (wave>>1)*16, cols ncol = blockIdx*32 + (wave&1)*16.
// Five accumulators per wave:
//   accd = c @ W_d^T              (K=512)
//   accf/acci/acco/accc = h @ W_all^T + x_t @ U^T   (K=512 + K=256)
// then the full cell update in registers.
__global__ void tl_step_kernel(const bf16_t* __restrict__ h_bf_in,
                               const bf16_t* __restrict__ c_bf_in,
                               const bf16_t* __restrict__ x_bf,
                               const bf16_t* __restrict__ Wall_bf,
                               const bf16_t* __restrict__ U_bf,
                               const bf16_t* __restrict__ Wd_bf,
                               const float* __restrict__ Wd_b,
                               const float* __restrict__ bias4,
                               const float* __restrict__ decay,
                               float* __restrict__ out,     // [B][S][H]
                               float* __restrict__ hf_out,  // [B][H]
                               float* __restrict__ cf_out,  // [B][H]
                               bf16_t* __restrict__ h_bf_out,
                               bf16_t* __restrict__ c_bf_out,
                               float* __restrict__ c_f32,   // read old / write new
                               int t) {
  const int lane = threadIdx.x & 31;
  const int wave = threadIdx.x >> 5;
  const int m0   = (wave >> 1) * 16;
  const int ncol = blockIdx.x * 32 + (wave & 1) * 16;

  v8f accd = {}, accf = {}, acci = {}, acco = {}, accc = {};

  // c @ W_d^T  (K = 512)
  for (int kk = 0; kk < HH; kk += 32) {
    v16bf a = load_a_frag(c_bf_in, HH, m0, kk, lane);
    v16bf b = load_b_frag(Wd_bf, HH, ncol, kk, lane);
    accd = wmma_bf16(a, b, accd);
  }
  // h @ W_all^T  (K = 512), all four gates off one shared A-fragment
  for (int kk = 0; kk < HH; kk += 32) {
    v16bf a  = load_a_frag(h_bf_in, HH, m0, kk, lane);
    v16bf b0 = load_b_frag(Wall_bf, HH, ncol + 0 * HH, kk, lane);
    v16bf b1 = load_b_frag(Wall_bf, HH, ncol + 1 * HH, kk, lane);
    v16bf b2 = load_b_frag(Wall_bf, HH, ncol + 2 * HH, kk, lane);
    v16bf b3 = load_b_frag(Wall_bf, HH, ncol + 3 * HH, kk, lane);
    accf = wmma_bf16(a, b0, accf);
    acci = wmma_bf16(a, b1, acci);
    acco = wmma_bf16(a, b2, acco);
    accc = wmma_bf16(a, b3, accc);
  }
  // x_t @ U^T  (K = 256); x is [B][S][E] -> row stride S*E, offset t*E
  {
    const bf16_t* xt = x_bf + (long)t * EE;
    for (int kk = 0; kk < EE; kk += 32) {
      v16bf a  = load_a_frag(xt, (long)SS * EE, m0, kk, lane);
      v16bf b0 = load_b_frag(U_bf, EE, ncol + 0 * HH, kk, lane);
      v16bf b1 = load_b_frag(U_bf, EE, ncol + 1 * HH, kk, lane);
      v16bf b2 = load_b_frag(U_bf, EE, ncol + 2 * HH, kk, lane);
      v16bf b3 = load_b_frag(U_bf, EE, ncol + 3 * HH, kk, lane);
      accf = wmma_bf16(a, b0, accf);
      acci = wmma_bf16(a, b1, acci);
      acco = wmma_bf16(a, b2, acco);
      accc = wmma_bf16(a, b3, accc);
    }
  }

  const int col   = ncol + (lane & 15);
  const int rbase = m0 + ((lane >> 4) << 3);
  const float bd_b = Wd_b[col];
  const float bf_b = bias4[0 * HH + col];
  const float bi_b = bias4[1 * HH + col];
  const float bo_b = bias4[2 * HH + col];
  const float bc_b = bias4[3 * HH + col];
#pragma unroll
  for (int r = 0; r < 8; ++r) {
    const int row = rbase + r;
    const float cs1  = tanhf(accd[r] + bd_b);
    const float cold = c_f32[row * HH + col];
    const float dec  = decay[row * SS + t];
    const float cadj = (cold - cs1) + cs1 * dec;
    const float fg = sigmoidf_(accf[r] + bf_b);
    const float ig = sigmoidf_(acci[r] + bi_b);
    const float og = sigmoidf_(acco[r] + bo_b);
    const float ct = tanhf(accc[r] + bc_b);
    const float cnew = fg * cadj + ig * ct;
    const float hnew = og * tanhf(cnew);
    out[((long)row * SS + t) * HH + col] = hnew;
    hf_out[row * HH + col]   = hnew;
    cf_out[row * HH + col]   = cnew;
    h_bf_out[row * HH + col] = (bf16_t)hnew;
    c_bf_out[row * HH + col] = (bf16_t)cnew;
    c_f32[row * HH + col]    = cnew;   // own tile only: read-before-write
  }
}

// ---------------------------------------------------------------- host -----
extern "C" void kernel_launch(void* const* d_in, const int* in_sizes, int n_in,
                              void* d_out, int out_size, void* d_ws, size_t ws_size,
                              hipStream_t stream) {
  const float* inputs  = (const float*)d_in[0];
  const float* ts      = (const float*)d_in[1];
  const float* W_all_w = (const float*)d_in[2];
  const float* W_all_b = (const float*)d_in[3];
  const float* U_all_w = (const float*)d_in[4];
  const float* U_all_b = (const float*)d_in[5];
  const float* W_d_w   = (const float*)d_in[6];
  const float* W_d_b   = (const float*)d_in[7];

  float* out = (float*)d_out;                     // [B][S][H]
  float* hf  = out + (long)BB * SS * HH;          // [B][H]
  float* cf  = hf + (long)BB * HH;                // [B][H]

  // Carve scratch (256B aligned chunks).
  char* ws = (char*)d_ws;
  auto carve = [&](size_t bytes) -> char* {
    char* p = ws;
    ws += (bytes + 255) & ~(size_t)255;
    return p;
  };
  bf16_t* x_bf    = (bf16_t*)carve((size_t)BB * SS * EE * sizeof(bf16_t));
  bf16_t* Wall_bf = (bf16_t*)carve((size_t)G4 * HH * sizeof(bf16_t));
  bf16_t* U_bf    = (bf16_t*)carve((size_t)G4 * EE * sizeof(bf16_t));
  bf16_t* Wd_bf   = (bf16_t*)carve((size_t)HH * HH * sizeof(bf16_t));
  float*  bias4   = (float*)carve((size_t)G4 * sizeof(float));
  float*  decay   = (float*)carve((size_t)BB * SS * sizeof(float));
  bf16_t* h_ping  = (bf16_t*)carve((size_t)BB * HH * sizeof(bf16_t));
  bf16_t* h_pong  = (bf16_t*)carve((size_t)BB * HH * sizeof(bf16_t));
  bf16_t* c_ping  = (bf16_t*)carve((size_t)BB * HH * sizeof(bf16_t));
  bf16_t* c_pong  = (bf16_t*)carve((size_t)BB * HH * sizeof(bf16_t));
  float*  c_f32   = (float*)carve((size_t)BB * HH * sizeof(float));
  (void)ws_size; (void)in_sizes; (void)n_in; (void)out_size;

  tl_setup_kernel<<<2048, 256, 0, stream>>>(
      inputs, ts, W_all_w, W_all_b, U_all_w, U_all_b, W_d_w,
      x_bf, Wall_bf, U_bf, Wd_bf, bias4, decay,
      h_ping, h_pong, c_ping, c_pong, c_f32);

  for (int t = 0; t < SS; ++t) {
    const bf16_t* h_in  = (t & 1) ? h_pong : h_ping;
    bf16_t*       h_out = (t & 1) ? h_ping : h_pong;
    const bf16_t* c_in  = (t & 1) ? c_pong : c_ping;
    bf16_t*       c_out = (t & 1) ? c_ping : c_pong;
    tl_step_kernel<<<HH / 32, 256, 0, stream>>>(
        h_in, c_in, x_bf, Wall_bf, U_bf, Wd_bf, W_d_b, bias4, decay,
        out, hf, cf, h_out, c_out, c_f32, t);
  }
}
